// RKdAngleBatchedMemoryEff_27479200759850
// MI455X (gfx1250) — matrix operand order repaired and checked by reading, not compiled, not run on hardware
//
#include <hip/hip_runtime.h>
#include <hip/hip_bf16.h>

typedef __attribute__((ext_vector_type(16))) _Float16 v16h;
typedef __attribute__((ext_vector_type(8)))  _Float16 v8h;
typedef __attribute__((ext_vector_type(8)))  float    v8f;

#define BB   4
#define NN   64
#define DD   128
#define NP   4096              // N*N pairs per batch
#define TILE_ROWS 32           // per block (2 wave-rows of 16)
#define TILE_COLS 64           // per block (4 wave-cols of 16)
#define GRID_X (NP / TILE_ROWS)   // 128
#define GRID_Y (NP / TILE_COLS)   // 64
#define NUM_PARTIALS (GRID_X * GRID_Y * BB)  // 32768

// ---------------------------------------------------------------------------
// Kernel 1: unit-normalized pairwise diffs for student & teacher, f32 -> f16.
// One 128-thread block per (b, i, j). LDS tree reduction for the norms.
// ---------------------------------------------------------------------------
__global__ void rkd_diff_kernel(const float* __restrict__ student,
                                const float* __restrict__ teacher,
                                _Float16* __restrict__ sd,
                                _Float16* __restrict__ td) {
    __shared__ float rs[DD];
    __shared__ float rt[DD];

    const int p  = blockIdx.x;        // 0 .. B*NP-1
    const int b  = p / NP;
    const int ij = p - b * NP;
    const int i  = ij >> 6;           // ij / 64
    const int j  = ij & 63;           // ij % 64
    const int t  = threadIdx.x;       // 0 .. 127 (D index)

    const float si = student[((size_t)(b * NN + i)) * DD + t];
    const float sj = student[((size_t)(b * NN + j)) * DD + t];
    const float ti = teacher[((size_t)(b * NN + i)) * DD + t];
    const float tj = teacher[((size_t)(b * NN + j)) * DD + t];
    const float ds = si - sj;
    const float dt = ti - tj;

    rs[t] = ds * ds;
    rt[t] = dt * dt;
    __syncthreads();
    #pragma unroll
    for (int off = DD / 2; off > 0; off >>= 1) {
        if (t < off) {
            rs[t] += rs[t + off];
            rt[t] += rt[t + off];
        }
        __syncthreads();
    }
    const float inv_s = 1.0f / fmaxf(sqrtf(rs[0]), 1e-12f);
    const float inv_t = 1.0f / fmaxf(sqrtf(rt[0]), 1e-12f);

    sd[(size_t)p * DD + t] = (_Float16)(ds * inv_s);
    td[(size_t)p * DD + t] = (_Float16)(dt * inv_t);
}

// ---------------------------------------------------------------------------
// Kernel 2: fused Gram (X X^T via WMMA f16->f32) + smooth-L1 epilogue.
// 256 threads = 8 waves; each wave owns one 16x16 output tile.
// Per ISA layouts (CDNA5 ISA):
//   A 16x32 f16: lane l (m=l&15, half=l>>4): elems 0..7  = K kbase..kbase+7,
//                elems 8..15 = K kbase+16..kbase+23, kbase = ks + 8*half.
//   B 32x16 f16: lane l (n=l&15, half=l>>4): same K order as A (elems 0..7
//                = K kbase..kbase+7, elems 8..15 = K kbase+16..kbase+23).
// ---------------------------------------------------------------------------
__global__ void rkd_gram_loss_kernel(const _Float16* __restrict__ sd,
                                     const _Float16* __restrict__ td,
                                     float* __restrict__ partials) {
    __shared__ float lred[256];

    const int b      = blockIdx.z;
    const int wave   = threadIdx.x >> 5;    // 0..7
    const int lane   = threadIdx.x & 31;
    const int half   = lane >> 4;
    const int m      = lane & 15;
    const int kt     = blockIdx.x * 2 + (wave & 1);   // row tile  [0,256)
    const int lt     = blockIdx.y * 4 + (wave >> 1);  // col tile  [0,256)

    const _Float16* Xs = sd + (size_t)b * ((size_t)NP * DD);
    const _Float16* Xt = td + (size_t)b * ((size_t)NP * DD);

    const int aRow = kt * 16 + m;
    const int bCol = lt * 16 + m;
    const _Float16* sA = Xs + (size_t)aRow * DD;
    const _Float16* sB = Xs + (size_t)bCol * DD;
    const _Float16* tA = Xt + (size_t)aRow * DD;
    const _Float16* tB = Xt + (size_t)bCol * DD;

    v8f accS = {};
    v8f accT = {};

    #pragma unroll
    for (int ks = 0; ks < DD; ks += 32) {
        const int kA = ks + 8 * half;    // A: two 8-elem chunks
        const int kB = ks + 16 * half;   // unused: B loads two 8-elem chunks at kA, like A

        v8h  sa0 = *(const v8h*)(sA + kA);
        v8h  sa1 = *(const v8h*)(sA + kA + 16);
        v16h aS  = __builtin_shufflevector(sa0, sa1,
                     0,1,2,3,4,5,6,7,8,9,10,11,12,13,14,15);
        v16h bS  = __builtin_shufflevector(*(const v8h*)(sB + kA), *(const v8h*)(sB + kA + 16),
                     0,1,2,3,4,5,6,7,8,9,10,11,12,13,14,15);

        v8h  ta0 = *(const v8h*)(tA + kA);
        v8h  ta1 = *(const v8h*)(tA + kA + 16);
        v16h aT  = __builtin_shufflevector(ta0, ta1,
                     0,1,2,3,4,5,6,7,8,9,10,11,12,13,14,15);
        v16h bT  = __builtin_shufflevector(*(const v8h*)(tB + kA), *(const v8h*)(tB + kA + 16),
                     0,1,2,3,4,5,6,7,8,9,10,11,12,13,14,15);

        accS = __builtin_amdgcn_wmma_f32_16x16x32_f16(
                   false, aS, false, bS, (short)0, accS, false, false);
        accT = __builtin_amdgcn_wmma_f32_16x16x32_f16(
                   false, aT, false, bT, (short)0, accT, false, false);
    }

    // Smooth-L1 (beta = 1) epilogue on the 8 f32 elements this lane owns.
    float lsum = 0.0f;
    #pragma unroll
    for (int e = 0; e < 8; ++e) {
        const float diff = accS[e] - accT[e];
        const float ad   = fabsf(diff);
        lsum += (ad < 1.0f) ? (0.5f * diff * diff) : (ad - 0.5f);
    }

    // Block reduction (deterministic), one partial per block.
    lred[threadIdx.x] = lsum;
    __syncthreads();
    #pragma unroll
    for (int off = 128; off > 0; off >>= 1) {
        if ((int)threadIdx.x < off) lred[threadIdx.x] += lred[threadIdx.x + off];
        __syncthreads();
    }
    if (threadIdx.x == 0) {
        const int pid = (blockIdx.z * GRID_Y + blockIdx.y) * GRID_X + blockIdx.x;
        partials[pid] = lred[0];
    }
}

// ---------------------------------------------------------------------------
// Kernel 3: deterministic final reduction of the block partials.
// ---------------------------------------------------------------------------
__global__ void rkd_reduce_kernel(const float* __restrict__ partials,
                                  float* __restrict__ out) {
    __shared__ float lred[256];
    const int t = threadIdx.x;
    float s = 0.0f;
    for (int i = t; i < NUM_PARTIALS; i += 256) s += partials[i];
    lred[t] = s;
    __syncthreads();
    #pragma unroll
    for (int off = 128; off > 0; off >>= 1) {
        if (t < off) lred[t] += lred[t + off];
        __syncthreads();
    }
    if (t == 0) {
        const float inv_count = 1.0f / ((float)BB * (float)NP * (float)NP);
        out[0] = lred[0] * inv_count;
    }
}

extern "C" void kernel_launch(void* const* d_in, const int* in_sizes, int n_in,
                              void* d_out, int out_size, void* d_ws, size_t ws_size,
                              hipStream_t stream) {
    const float* student = (const float*)d_in[0];
    const float* teacher = (const float*)d_in[1];
    float* out = (float*)d_out;

    // Workspace layout:
    //   sd f16: B*NP*DD = 2,097,152 elems (4 MB)
    //   td f16: 4 MB
    //   partials f32: NUM_PARTIALS (128 KB)
    _Float16* sd = (_Float16*)d_ws;
    _Float16* td = sd + (size_t)BB * NP * DD;
    float* partials = (float*)(td + (size_t)BB * NP * DD);

    // Kernel 1: unit pairwise diffs (16384 blocks x 128 threads).
    rkd_diff_kernel<<<BB * NP, DD, 0, stream>>>(student, teacher, sd, td);

    // Kernel 2: WMMA Gram + fused smooth-L1 (128 x 64 x 4 blocks x 256 threads).
    dim3 g2(GRID_X, GRID_Y, BB);
    rkd_gram_loss_kernel<<<g2, 256, 0, stream>>>(sd, td, partials);

    // Kernel 3: final deterministic reduction.
    rkd_reduce_kernel<<<1, 256, 0, stream>>>(partials, out);
}
